// ExpertAdapter_13640816132847
// MI455X (gfx1250) — matrix operand order repaired and checked by reading, not compiled, not run on hardware
//
#include <hip/hip_runtime.h>

#define D_DIM 1024
#define F_DIM 512
#define E_NUM 8
#define T_TOK 16384

#define XS_STRIDE 1032   // 1024 + 8 pad (LDS bank-conflict avoidance, keeps 16B align)
#define HS_STRIDE 520    // 512 + 8 pad

typedef __attribute__((ext_vector_type(16))) __bf16        v16bf;
typedef __attribute__((ext_vector_type(8)))  float         v8f;
typedef __attribute__((ext_vector_type(4)))  unsigned int  u32x4;
typedef __attribute__((ext_vector_type(2)))  unsigned int  u32x2;
typedef __attribute__((ext_vector_type(4)))  float         f32x4;

union Frag {
  u32x4 q[2];
  v16bf v;
};

__device__ __forceinline__ unsigned int bf16_rne(float f) {
  unsigned int u = __float_as_uint(f);
  return (u + 0x7FFFu + ((u >> 16) & 1u)) >> 16;
}
__device__ __forceinline__ unsigned int pack_bf16(float lo, float hi) {
  return bf16_rne(lo) | (bf16_rne(hi) << 16);
}
__device__ __forceinline__ float gelu_exact(float x) {
  return 0.5f * x * (1.0f + erff(x * 0.70710678118654752f));
}
__device__ __forceinline__ v8f vzero8() {
  v8f z = {0.f, 0.f, 0.f, 0.f, 0.f, 0.f, 0.f, 0.f};
  return z;
}

// ---------------- prep: W1 [E,D,F] f32 -> W1t [E,F,D] bf16 ----------------
__global__ void transpose_w1_kernel(const float* __restrict__ W1,
                                    unsigned short* __restrict__ W1t) {
  int o = blockIdx.x * blockDim.x + threadIdx.x;   // output-linear over E*F*D
  int d  = o & (D_DIM - 1);
  int ef = o >> 10;
  int f  = ef & (F_DIM - 1);
  int e  = ef >> 9;
  float v = W1[(e * D_DIM + d) * F_DIM + f];
  W1t[o] = (unsigned short)bf16_rne(v);
}

// ---------------- prep: W2 [E,F,D] f32 -> W2t [E,D,F] bf16 ----------------
__global__ void transpose_w2_kernel(const float* __restrict__ W2,
                                    unsigned short* __restrict__ W2t) {
  int o = blockIdx.x * blockDim.x + threadIdx.x;   // output-linear over E*D*F
  int f  = o & (F_DIM - 1);
  int ed = o >> 9;
  int d  = ed & (D_DIM - 1);
  int e  = ed >> 10;
  float v = W2[(e * F_DIM + f) * D_DIM + d];
  W2t[o] = (unsigned short)bf16_rne(v);
}

// ---------------- router: softmax(x @ Wr + br) per token ----------------
__global__ void router_kernel(const float* __restrict__ x,
                              const float* __restrict__ Wr,
                              const float* __restrict__ br,
                              float* __restrict__ router) {
  int t = blockIdx.x * blockDim.x + threadIdx.x;
  if (t >= T_TOK) return;
  float acc[E_NUM];
#pragma unroll
  for (int e = 0; e < E_NUM; ++e) acc[e] = br[e];
  const float* xr = x + (size_t)t * D_DIM;
  for (int d = 0; d < D_DIM; ++d) {
    float xv = xr[d];
#pragma unroll
    for (int e = 0; e < E_NUM; ++e) acc[e] = fmaf(xv, Wr[d * E_NUM + e], acc[e]);
  }
  float m = acc[0];
#pragma unroll
  for (int e = 1; e < E_NUM; ++e) m = fmaxf(m, acc[e]);
  float s = 0.f;
#pragma unroll
  for (int e = 0; e < E_NUM; ++e) { acc[e] = expf(acc[e] - m); s += acc[e]; }
  float inv = 1.f / s;
#pragma unroll
  for (int e = 0; e < E_NUM; ++e) router[t * E_NUM + e] = acc[e] * inv;
}

// ---------------- fused MoE: 16 tokens per block, 8 waves ----------------
__launch_bounds__(256)
__global__ void moe_kernel(const float* __restrict__ x,
                           const unsigned short* __restrict__ W1t,  // [E,F,D] bf16
                           const float* __restrict__ b1,            // [E,F]
                           const unsigned short* __restrict__ W2t,  // [E,D,F] bf16
                           const float* __restrict__ b2,            // [E,D]
                           const float* __restrict__ router,        // [T,E]
                           float* __restrict__ out) {               // [T,D]
  __shared__ unsigned short xs[16 * XS_STRIDE];  // x tile, bf16 [token][d]
  __shared__ unsigned short hs[16 * HS_STRIDE];  // h tile, bf16 [token][f]
  __shared__ float rs[16 * E_NUM];

  const int tid   = threadIdx.x;
  const int tok0  = blockIdx.x * 16;
  const int wave  = tid >> 5;
  const int lane  = tid & 31;
  const int lmod  = lane & 15;   // token column / M-row within 16
  const int lhalf = lane >> 4;   // K-half selector

  // ---- stage x tile: f32 global -> bf16 LDS (16 x 1024) ----
  {
    const f32x4* xg = (const f32x4*)(x + (size_t)tok0 * D_DIM);
#pragma unroll
    for (int j = 0; j < 16; ++j) {
      int idx4 = tid + j * 256;          // 0..4095 float4s
      int row  = idx4 >> 8;              // 256 float4 per row
      int col4 = idx4 & 255;
      f32x4 v = xg[idx4];
      u32x2 p;
      p.x = pack_bf16(v.x, v.y);
      p.y = pack_bf16(v.z, v.w);
      *(u32x2*)&xs[row * XS_STRIDE + col4 * 4] = p;
    }
    if (tid < 16 * E_NUM) rs[tid] = router[tok0 * E_NUM + tid];
  }
  __syncthreads();

  v8f outacc[8];
#pragma unroll
  for (int i = 0; i < 8; ++i) outacc[i] = vzero8();

#pragma unroll 1
  for (int e = 0; e < E_NUM; ++e) {
    // ===== GEMM1: h^T = W1t[e] (F x D) * x^T (D x 16); wave owns 64 f-rows =====
    v8f c1[4];
#pragma unroll
    for (int i = 0; i < 4; ++i) c1[i] = vzero8();
    const unsigned short* w1e = W1t + (size_t)e * F_DIM * D_DIM;
#pragma unroll 1
    for (int kb = 0; kb < D_DIM; kb += 32) {
      Frag bf;
      const unsigned short* bp = &xs[lmod * XS_STRIDE + kb + lhalf * 8];
      bf.q[0] = *(const u32x4*)bp;
      bf.q[1] = *(const u32x4*)(bp + 16);
#pragma unroll
      for (int mt = 0; mt < 4; ++mt) {
        Frag af;
        const unsigned short* ap =
            w1e + (size_t)(wave * 64 + mt * 16 + lmod) * D_DIM + kb + lhalf * 8;
        af.q[0] = *(const u32x4*)ap;
        af.q[1] = *(const u32x4*)(ap + 16);
        c1[mt] = __builtin_amdgcn_wmma_f32_16x16x32_bf16(
            false, af.v, false, bf.v, (short)0, c1[mt], false, false);
      }
    }
    __syncthreads();  // all waves done reading hs from previous expert
    // bias + exact gelu, pack bf16, write h row-major [token][f]
#pragma unroll
    for (int mt = 0; mt < 4; ++mt) {
      int fbase = wave * 64 + mt * 16 + lhalf * 8;
      const float* bb = b1 + e * F_DIM + fbase;
      float g[8];
#pragma unroll
      for (int i = 0; i < 8; ++i) g[i] = gelu_exact(c1[mt][i] + bb[i]);
      u32x4 pk;
      pk.x = pack_bf16(g[0], g[1]);
      pk.y = pack_bf16(g[2], g[3]);
      pk.z = pack_bf16(g[4], g[5]);
      pk.w = pack_bf16(g[6], g[7]);
      *(u32x4*)&hs[lmod * HS_STRIDE + fbase] = pk;
    }
    __syncthreads();

    // ===== GEMM2: y^T = W2t[e] (D x F) * h^T (F x 16); wave owns 128 d-rows =====
    v8f c2[8];
#pragma unroll
    for (int i = 0; i < 8; ++i) c2[i] = vzero8();
    const unsigned short* w2e = W2t + (size_t)e * D_DIM * F_DIM;
#pragma unroll 1
    for (int kb = 0; kb < F_DIM; kb += 32) {
      Frag bf;
      const unsigned short* bp = &hs[lmod * HS_STRIDE + kb + lhalf * 8];
      bf.q[0] = *(const u32x4*)bp;
      bf.q[1] = *(const u32x4*)(bp + 16);
#pragma unroll
      for (int mt = 0; mt < 8; ++mt) {
        Frag af;
        const unsigned short* ap =
            w2e + (size_t)(wave * 128 + mt * 16 + lmod) * F_DIM + kb + lhalf * 8;
        af.q[0] = *(const u32x4*)ap;
        af.q[1] = *(const u32x4*)(ap + 16);
        c2[mt] = __builtin_amdgcn_wmma_f32_16x16x32_bf16(
            false, af.v, false, bf.v, (short)0, c2[mt], false, false);
      }
    }
    // router-weighted accumulate (each lane holds exactly one token = lmod)
    float r = rs[lmod * E_NUM + e];
#pragma unroll
    for (int mt = 0; mt < 8; ++mt) {
      int dbase = wave * 128 + mt * 16 + lhalf * 8;
      const float* bb = b2 + e * D_DIM + dbase;
#pragma unroll
      for (int i = 0; i < 8; ++i)
        outacc[mt][i] += r * (c2[mt][i] + bb[i]);
    }
    // next iteration's first __syncthreads() protects hs before rewrite
  }

  // ---- write out [token][d] ----
#pragma unroll
  for (int mt = 0; mt < 8; ++mt) {
    int dbase = wave * 128 + mt * 16 + lhalf * 8;
    float* op = out + (size_t)(tok0 + lmod) * D_DIM + dbase;
    f32x4 lo, hi;
    lo.x = outacc[mt][0]; lo.y = outacc[mt][1]; lo.z = outacc[mt][2]; lo.w = outacc[mt][3];
    hi.x = outacc[mt][4]; hi.y = outacc[mt][5]; hi.z = outacc[mt][6]; hi.w = outacc[mt][7];
    *(f32x4*)op       = lo;
    *(f32x4*)(op + 4) = hi;
  }
}

extern "C" void kernel_launch(void* const* d_in, const int* in_sizes, int n_in,
                              void* d_out, int out_size, void* d_ws, size_t ws_size,
                              hipStream_t stream) {
  (void)in_sizes; (void)n_in; (void)out_size; (void)ws_size;
  const float* x  = (const float*)d_in[0];
  const float* W1 = (const float*)d_in[1];
  const float* b1 = (const float*)d_in[2];
  const float* W2 = (const float*)d_in[3];
  const float* b2 = (const float*)d_in[4];
  const float* Wr = (const float*)d_in[5];
  const float* br = (const float*)d_in[6];
  float* out = (float*)d_out;

  char* ws = (char*)d_ws;
  const size_t wbytes = (size_t)E_NUM * F_DIM * D_DIM * 2;  // 8 MB each
  unsigned short* W1t = (unsigned short*)ws;
  unsigned short* W2t = (unsigned short*)(ws + wbytes);
  float* router = (float*)(ws + 2 * wbytes);                // 512 KB

  transpose_w1_kernel<<<(E_NUM * F_DIM * D_DIM) / 256, 256, 0, stream>>>(W1, W1t);
  transpose_w2_kernel<<<(E_NUM * D_DIM * F_DIM) / 256, 256, 0, stream>>>(W2, W2t);
  router_kernel<<<T_TOK / 256, 256, 0, stream>>>(x, Wr, br, router);
  moe_kernel<<<T_TOK / 16, 256, 0, stream>>>(x, W1t, b1, W2t, b2, router, out);
}
